// CrossAttention_66571993088196
// MI455X (gfx1250) — matrix operand order repaired and checked by reading, not compiled, not run on hardware
//
#include <hip/hip_runtime.h>
#include <cmath>

typedef __attribute__((ext_vector_type(16))) _Float16 v16h;
typedef __attribute__((ext_vector_type(8)))  float    v8f;

#define WMMA_F16(A, B, Cacc) \
  __builtin_amdgcn_wmma_f32_16x16x32_f16(false, (A), false, (B), (short)0, (Cacc), false, false)

__device__ __forceinline__ v16h ldfrag(const _Float16* __restrict__ base,
                                       int mt, int kt, int KT, int lane) {
  return *(const v16h*)(base + (((size_t)(mt * KT + kt)) * 32 + (size_t)lane) * 16);
}

// -------------------------------------------------------------------------
// prep: fold BN into A1/P1, convert GEMM weights to f16 in A-fragment layout
// A-frag element j (lane half h): k = h*16 + j
// -------------------------------------------------------------------------
__device__ void swz(const float* __restrict__ W, int M, int K,
                    const float* __restrict__ g, const float* __restrict__ vv,
                    _Float16* __restrict__ out, int tid, int nth) {
  const int MT = M >> 4, KT = K >> 5, total = MT * KT * 512;
  for (int e = tid; e < total; e += nth) {
    int j = e & 15, lane = (e >> 4) & 31, tile = e >> 9;
    int kt = tile % KT, mt = tile / KT;
    int h = lane >> 4;
    int m = mt * 16 + (lane & 15);
    int k = kt * 32 + h * 16 + j;
    float w = W[(size_t)m * K + k];
    if (g) w *= g[m] * rsqrtf(vv[m] + 1e-5f);
    out[e] = (_Float16)w;
  }
}

__global__ __launch_bounds__(256) void prep_kernel(
    const float* __restrict__ W_start, const float* __restrict__ W_key,
    const float* __restrict__ W_end,
    const float* __restrict__ P1, const float* __restrict__ pb1,
    const float* __restrict__ bn1_g, const float* __restrict__ bn1_b,
    const float* __restrict__ bn1_m, const float* __restrict__ bn1_v,
    const float* __restrict__ P2,
    const float* __restrict__ A1, const float* __restrict__ ab1,
    const float* __restrict__ bn2_g, const float* __restrict__ bn2_b,
    const float* __restrict__ bn2_m, const float* __restrict__ bn2_v,
    const float* __restrict__ A2,
    _Float16* __restrict__ fWs, _Float16* __restrict__ fWk,
    _Float16* __restrict__ fWe, _Float16* __restrict__ fP2,
    _Float16* __restrict__ fA1, _Float16* __restrict__ fA2,
    float* __restrict__ bias1p, float* __restrict__ P1f, float* __restrict__ pb1f) {
  const int tid = blockIdx.x * blockDim.x + threadIdx.x;
  const int nth = gridDim.x * blockDim.x;
  swz(W_start, 64, 64, nullptr, nullptr, fWs, tid, nth);
  swz(W_key,   64, 64, nullptr, nullptr, fWk, tid, nth);
  swz(W_end,   64, 64, nullptr, nullptr, fWe, tid, nth);
  swz(P2,      64, 64, nullptr, nullptr, fP2, tid, nth);
  swz(A1,     256, 64, bn2_g,   bn2_v,   fA1, tid, nth);
  swz(A2,     64, 256, nullptr, nullptr, fA2, tid, nth);
  for (int m = tid; m < 256; m += nth) {
    float s = bn2_g[m] * rsqrtf(bn2_v[m] + 1e-5f);
    bias1p[m] = ab1[m] * s + bn2_b[m] - bn2_m[m] * s;
  }
  for (int m = tid; m < 64; m += nth) {
    float s = bn1_g[m] * rsqrtf(bn1_v[m] + 1e-5f);
    pb1f[m] = pb1[m] * s + bn1_b[m] - bn1_m[m] * s;
    #pragma unroll
    for (int d = 0; d < 3; ++d) P1f[m * 3 + d] = P1[m * 3 + d] * s;
  }
}

// -------------------------------------------------------------------------
// kNN: block = 256 query points of one batch; all 4096 points in LDS (48KB)
// -------------------------------------------------------------------------
__global__ __launch_bounds__(256) void knn_kernel(const float* __restrict__ x,
                                                  int* __restrict__ idx) {
  __shared__ float sx[4096], sy[4096], sz[4096];
  const int b = blockIdx.x >> 4;          // 16 blocks per batch
  const int tile = blockIdx.x & 15;
  const float* xb = x + (size_t)b * 3 * 4096;
  for (int i = threadIdx.x; i < 4096; i += 256) {
    sx[i] = xb[i]; sy[i] = xb[4096 + i]; sz[i] = xb[8192 + i];
  }
  __syncthreads();
  const int q = tile * 256 + threadIdx.x;
  const float qx = sx[q], qy = sy[q], qz = sz[q];
  float bd[16]; int bi[16];
  #pragma unroll
  for (int t = 0; t < 16; ++t) { bd[t] = 3.0e38f; bi[t] = 0; }
  for (int j = 0; j < 4096; ++j) {
    float dx = qx - sx[j], dy = qy - sy[j], dz = qz - sz[j];
    float d = dx * dx + dy * dy + dz * dz;
    if (d < bd[15]) {
      float cd = d; int ci = j;
      #pragma unroll
      for (int t = 0; t < 16; ++t) {
        if (cd < bd[t]) {
          float td = bd[t]; int ti = bi[t];
          bd[t] = cd; bi[t] = ci; cd = td; ci = ti;
        }
      }
    }
  }
  int* o = idx + ((size_t)b * 4096 + q) * 16;
  #pragma unroll
  for (int t = 0; t < 16; ++t) o[t] = bi[t];
}

// -------------------------------------------------------------------------
// pointwise: y1=Ws*y+b (value), key=Wk*y1+b, query=Wq*x+b. 1 wave / 16 pts.
// -------------------------------------------------------------------------
__global__ __launch_bounds__(256) void pointwise_kernel(
    const float* __restrict__ x, const float* __restrict__ y,
    const _Float16* __restrict__ fWs, const float* __restrict__ b_start,
    const _Float16* __restrict__ fWk, const float* __restrict__ b_key,
    const float* __restrict__ W_query, const float* __restrict__ b_query,
    float* __restrict__ value, _Float16* __restrict__ keyh,
    float* __restrict__ query) {
  __shared__ float lds[8 * 64 * 16];
  const int lane = threadIdx.x & 31, wv = threadIdx.x >> 5;
  const int nl = lane & 15, h = lane >> 4;
  const int tile = blockIdx.x * 8 + wv;           // 1024 tiles
  const int b = tile >> 8;
  const int n = ((tile & 255) << 4) + nl;
  float* T = lds + wv * 1024;
  const size_t pt = (size_t)b * 4096 + n;

  // Y as B fragments: k = h*16 + j  (channel contiguous per lane)
  v16h yf0, yf1;
  #pragma unroll
  for (int j = 0; j < 16; ++j) {
    int c = h * 16 + j;
    yf0[j] = (_Float16)y[((size_t)b * 64 + c) * 4096 + n];
    yf1[j] = (_Float16)y[((size_t)b * 64 + c + 32) * 4096 + n];
  }
  #pragma unroll
  for (int mt = 0; mt < 4; ++mt) {
    v8f acc;
    #pragma unroll
    for (int r = 0; r < 8; ++r) acc[r] = b_start[mt * 16 + h * 8 + r];
    acc = WMMA_F16(ldfrag(fWs, mt, 0, 2, lane), yf0, acc);
    acc = WMMA_F16(ldfrag(fWs, mt, 1, 2, lane), yf1, acc);
    #pragma unroll
    for (int r = 0; r < 8; ++r) {
      int c = mt * 16 + h * 8 + r;
      value[pt * 64 + c] = acc[r];
      T[c * 16 + nl] = acc[r];
    }
  }
  // D->B relayout via LDS, then key = Wk*y1
  v16h g0, g1;
  #pragma unroll
  for (int j = 0; j < 16; ++j) {
    int c = h * 16 + j;
    g0[j] = (_Float16)T[c * 16 + nl];
    g1[j] = (_Float16)T[(c + 32) * 16 + nl];
  }
  #pragma unroll
  for (int mt = 0; mt < 4; ++mt) {
    v8f acc;
    #pragma unroll
    for (int r = 0; r < 8; ++r) acc[r] = b_key[mt * 16 + h * 8 + r];
    acc = WMMA_F16(ldfrag(fWk, mt, 0, 2, lane), g0, acc);
    acc = WMMA_F16(ldfrag(fWk, mt, 1, 2, lane), g1, acc);
    #pragma unroll
    for (int r = 0; r < 8; ++r)
      keyh[pt * 64 + mt * 16 + h * 8 + r] = (_Float16)acc[r];
  }
  // query = Wq(64x3)*x + b  (VALU; K=3 too small for WMMA)
  float x0 = x[((size_t)b * 3 + 0) * 4096 + n];
  float x1 = x[((size_t)b * 3 + 1) * 4096 + n];
  float x2 = x[((size_t)b * 3 + 2) * 4096 + n];
  #pragma unroll
  for (int mt = 0; mt < 4; ++mt)
    #pragma unroll
    for (int r = 0; r < 8; ++r) {
      int c = mt * 16 + h * 8 + r;
      query[pt * 64 + c] = b_query[c] + W_query[c * 3] * x0 +
                           W_query[c * 3 + 1] * x1 + W_query[c * 3 + 2] * x2;
    }
}

// -------------------------------------------------------------------------
// fused attention: 1 wave / point; neighbors (K=16) = WMMA N dimension
// -------------------------------------------------------------------------
__global__ __launch_bounds__(256) void fused_attn_kernel(
    const float* __restrict__ x, const int* __restrict__ idx,
    const float* __restrict__ query, const _Float16* __restrict__ keyh,
    const float* __restrict__ value,
    const float* __restrict__ P1f, const float* __restrict__ pb1f,
    const _Float16* __restrict__ fP2, const float* __restrict__ pb2,
    const _Float16* __restrict__ fA1, const float* __restrict__ bias1p,
    const _Float16* __restrict__ fA2, const float* __restrict__ ab2,
    float* __restrict__ agg) {
  __shared__ float peS[8 * 64 * 16];   // 32KB
  __shared__ float hS[8 * 32 * 16];    // 16KB
  const int lane = threadIdx.x & 31, wv = threadIdx.x >> 5;
  const int nl = lane & 15, h = lane >> 4;
  const int pt = blockIdx.x * 8 + wv;              // b*4096+n
  const int b = pt >> 12, n = pt & 4095;
  float* peT = peS + wv * 1024;
  float* hT  = hS  + wv * 512;

  const int nbr = idx[(size_t)pt * 16 + nl];
  float pr0 = x[((size_t)b * 3 + 0) * 4096 + n] - x[((size_t)b * 3 + 0) * 4096 + nbr];
  float pr1 = x[((size_t)b * 3 + 1) * 4096 + n] - x[((size_t)b * 3 + 1) * 4096 + nbr];
  float pr2 = x[((size_t)b * 3 + 2) * 4096 + n] - x[((size_t)b * 3 + 2) * 4096 + nbr];

  // h1 = relu(bn-folded P1 * pos_rel), built directly in B-fragment layout
  v16h h1f0, h1f1;
  #pragma unroll
  for (int j = 0; j < 16; ++j) {
    int c = h * 16 + j;
    float v0 = pb1f[c] + P1f[c * 3] * pr0 + P1f[c * 3 + 1] * pr1 + P1f[c * 3 + 2] * pr2;
    int c1 = c + 32;
    float v1 = pb1f[c1] + P1f[c1 * 3] * pr0 + P1f[c1 * 3 + 1] * pr1 + P1f[c1 * 3 + 2] * pr2;
    h1f0[j] = (_Float16)fmaxf(v0, 0.f);
    h1f1[j] = (_Float16)fmaxf(v1, 0.f);
  }
  // pe = P2 * h1 + pb2
  #pragma unroll
  for (int mt = 0; mt < 4; ++mt) {
    v8f acc;
    #pragma unroll
    for (int r = 0; r < 8; ++r) acc[r] = pb2[mt * 16 + h * 8 + r];
    acc = WMMA_F16(ldfrag(fP2, mt, 0, 2, lane), h1f0, acc);
    acc = WMMA_F16(ldfrag(fP2, mt, 1, 2, lane), h1f1, acc);
    #pragma unroll
    for (int r = 0; r < 8; ++r) peT[(mt * 16 + h * 8 + r) * 16 + nl] = acc[r];
  }
  // X = qk_rel + pe as B fragments
  const float*    qp = query + (size_t)pt * 64;
  const _Float16* kp = keyh + ((size_t)b * 4096 + nbr) * 64;
  v16h xf0, xf1;
  #pragma unroll
  for (int j = 0; j < 16; ++j) {
    int c = h * 16 + j;
    xf0[j] = (_Float16)(qp[c] - (float)kp[c] + peT[c * 16 + nl]);
    int c1 = c + 32;
    xf1[j] = (_Float16)(qp[c1] - (float)kp[c1] + peT[c1 * 16 + nl]);
  }
  // attn = A2 * relu(bn-folded A1 * X), accumulated over 8 chunks of 32 rows
  v8f at[4];
  #pragma unroll
  for (int mt = 0; mt < 4; ++mt)
    #pragma unroll
    for (int r = 0; r < 8; ++r) at[mt][r] = ab2[mt * 16 + h * 8 + r];

  #pragma unroll
  for (int ch = 0; ch < 8; ++ch) {
    v8f t0, t1;
    #pragma unroll
    for (int r = 0; r < 8; ++r) {
      t0[r] = bias1p[ch * 32 + h * 8 + r];
      t1[r] = bias1p[ch * 32 + 16 + h * 8 + r];
    }
    t0 = WMMA_F16(ldfrag(fA1, 2 * ch, 0, 2, lane), xf0, t0);
    t0 = WMMA_F16(ldfrag(fA1, 2 * ch, 1, 2, lane), xf1, t0);
    t1 = WMMA_F16(ldfrag(fA1, 2 * ch + 1, 0, 2, lane), xf0, t1);
    t1 = WMMA_F16(ldfrag(fA1, 2 * ch + 1, 1, 2, lane), xf1, t1);
    #pragma unroll
    for (int r = 0; r < 8; ++r) {
      hT[(h * 8 + r) * 16 + nl]        = fmaxf(t0[r], 0.f);
      hT[(16 + h * 8 + r) * 16 + nl]   = fmaxf(t1[r], 0.f);
    }
    v16h hf;
    #pragma unroll
    for (int j = 0; j < 16; ++j) hf[j] = (_Float16)hT[(h * 16 + j) * 16 + nl];
    #pragma unroll
    for (int mt = 0; mt < 4; ++mt)
      at[mt] = WMMA_F16(ldfrag(fA2, mt, ch, 8, lane), hf, at[mt]);
  }
  // softmax over the 16 neighbor lanes (per channel), then agg
  const float* vp = value + (size_t)pt * 64;
  #pragma unroll
  for (int mt = 0; mt < 4; ++mt)
    #pragma unroll
    for (int r = 0; r < 8; ++r) {
      float v = at[mt][r];
      float m = v;
      #pragma unroll
      for (int msk = 1; msk < 16; msk <<= 1) m = fmaxf(m, __shfl_xor(m, msk, 32));
      float e = __expf(v - m);
      float s = e;
      #pragma unroll
      for (int msk = 1; msk < 16; msk <<= 1) s += __shfl_xor(s, msk, 32);
      int c = mt * 16 + h * 8 + r;
      float p = (e / s) * (vp[c] + peT[c * 16 + nl]);
      #pragma unroll
      for (int msk = 1; msk < 16; msk <<= 1) p += __shfl_xor(p, msk, 32);
      if (nl == 0) agg[(size_t)pt * 64 + c] = p;
    }
}

// -------------------------------------------------------------------------
// final: out = W_end*agg + b_end + identity
// -------------------------------------------------------------------------
__global__ __launch_bounds__(256) void final_kernel(
    const float* __restrict__ y, const float* __restrict__ agg,
    const _Float16* __restrict__ fWe, const float* __restrict__ b_end,
    float* __restrict__ out) {
  const int lane = threadIdx.x & 31, wv = threadIdx.x >> 5;
  const int nl = lane & 15, h = lane >> 4;
  const int tile = blockIdx.x * 8 + wv;
  const int b = tile >> 8;
  const int n = ((tile & 255) << 4) + nl;
  const float* ap = agg + ((size_t)b * 4096 + n) * 64;
  v16h af0, af1;
  #pragma unroll
  for (int j = 0; j < 16; ++j) {
    int c = h * 16 + j;
    af0[j] = (_Float16)ap[c];
    af1[j] = (_Float16)ap[c + 32];
  }
  #pragma unroll
  for (int mt = 0; mt < 4; ++mt) {
    v8f acc;
    #pragma unroll
    for (int r = 0; r < 8; ++r) acc[r] = b_end[mt * 16 + h * 8 + r];
    acc = WMMA_F16(ldfrag(fWe, mt, 0, 2, lane), af0, acc);
    acc = WMMA_F16(ldfrag(fWe, mt, 1, 2, lane), af1, acc);
    #pragma unroll
    for (int r = 0; r < 8; ++r) {
      int c = mt * 16 + h * 8 + r;
      size_t o = ((size_t)b * 64 + c) * 4096 + n;
      out[o] = acc[r] + y[o];
    }
  }
}

// -------------------------------------------------------------------------
extern "C" void kernel_launch(void* const* d_in, const int* in_sizes, int n_in,
                              void* d_out, int out_size, void* d_ws, size_t ws_size,
                              hipStream_t stream) {
  (void)in_sizes; (void)n_in; (void)out_size; (void)ws_size;
  const float* x       = (const float*)d_in[0];
  const float* y       = (const float*)d_in[1];
  const float* W_start = (const float*)d_in[2];
  const float* b_start = (const float*)d_in[3];
  const float* W_key   = (const float*)d_in[4];
  const float* b_key   = (const float*)d_in[5];
  const float* W_query = (const float*)d_in[6];
  const float* b_query = (const float*)d_in[7];
  const float* P1      = (const float*)d_in[8];
  const float* pb1     = (const float*)d_in[9];
  const float* bn1_g   = (const float*)d_in[10];
  const float* bn1_b   = (const float*)d_in[11];
  const float* bn1_m   = (const float*)d_in[12];
  const float* bn1_v   = (const float*)d_in[13];
  const float* P2      = (const float*)d_in[14];
  const float* pb2     = (const float*)d_in[15];
  const float* A1      = (const float*)d_in[16];
  const float* ab1     = (const float*)d_in[17];
  const float* bn2_g   = (const float*)d_in[18];
  const float* bn2_b   = (const float*)d_in[19];
  const float* bn2_m   = (const float*)d_in[20];
  const float* bn2_v   = (const float*)d_in[21];
  const float* A2      = (const float*)d_in[22];
  const float* ab2     = (const float*)d_in[23];
  const float* W_end   = (const float*)d_in[24];
  const float* b_end   = (const float*)d_in[25];
  float* out = (float*)d_out;

  constexpr size_t B = 4, N = 4096;
  size_t off = 0;
  auto alloc = [&](size_t bytes) -> void* {
    void* p = (char*)d_ws + off;
    off += (bytes + 255) & ~(size_t)255;
    return p;
  };
  _Float16* fWs   = (_Float16*)alloc(4096 * 2);
  _Float16* fWk   = (_Float16*)alloc(4096 * 2);
  _Float16* fWe   = (_Float16*)alloc(4096 * 2);
  _Float16* fP2   = (_Float16*)alloc(4096 * 2);
  _Float16* fA1   = (_Float16*)alloc(16384 * 2);
  _Float16* fA2   = (_Float16*)alloc(16384 * 2);
  float*    bias1p= (float*)alloc(256 * 4);
  float*    P1f   = (float*)alloc(64 * 3 * 4);
  float*    pb1f  = (float*)alloc(64 * 4);
  int*      idx   = (int*)alloc(B * N * 16 * 4);
  float*    value = (float*)alloc(B * N * 64 * 4);
  _Float16* keyh  = (_Float16*)alloc(B * N * 64 * 2);
  float*    query = (float*)alloc(B * N * 64 * 4);
  float*    aggb  = (float*)alloc(B * N * 64 * 4);

  prep_kernel<<<32, 256, 0, stream>>>(W_start, W_key, W_end, P1, pb1,
      bn1_g, bn1_b, bn1_m, bn1_v, P2, A1, ab1, bn2_g, bn2_b, bn2_m, bn2_v, A2,
      fWs, fWk, fWe, fP2, fA1, fA2, bias1p, P1f, pb1f);
  knn_kernel<<<64, 256, 0, stream>>>(x, idx);
  pointwise_kernel<<<128, 256, 0, stream>>>(x, y, fWs, b_start, fWk, b_key,
      W_query, b_query, value, keyh, query);
  fused_attn_kernel<<<2048, 256, 0, stream>>>(x, idx, query, keyh, value,
      P1f, pb1f, fP2, pb2, fA1, bias1p, fA2, ab2, aggb);
  final_kernel<<<128, 256, 0, stream>>>(y, aggb, fWe, b_end, out);
}
